// Scaled_Dot_Product_Attention_5257039971003
// MI455X (gfx1250) — matrix operand order repaired and checked by reading, not compiled, not run on hardware
//
#include <hip/hip_runtime.h>

typedef __attribute__((ext_vector_type(16))) _Float16 v16h;
typedef __attribute__((ext_vector_type(8)))  _Float16 v8h;
typedef __attribute__((ext_vector_type(4)))  _Float16 v4h;
typedef __attribute__((ext_vector_type(8)))  float    v8f;
typedef __attribute__((ext_vector_type(4)))  float    f32x4;

constexpr int Bn     = 8;     // batches
constexpr int Nn     = 4096;  // sequence length
constexpr int Dn     = 64;    // head dim
constexpr int QTILE  = 128;   // query rows per block (8 waves x 16 queries)
constexpr int KTILE  = 64;    // key rows per iteration
constexpr int NTILES = Nn / KTILE;

__device__ __forceinline__ v8f wmma_f16(v16h a, v16h b, v8f c) {
  // v_wmma_f32_16x16x32_f16: (neg_a, A, neg_b, B, c_mod, C, reuse_a, reuse_b)
  return __builtin_amdgcn_wmma_f32_16x16x32_f16(false, a, false, b, (short)0, c, false, false);
}

// A-layout (16x32 f16) fragment from a row-major f16 row in LDS:
// elements 0..7 = cols c0..c0+7, elements 8..15 = cols c0+8..c0+15
__device__ __forceinline__ v16h lds_afrag(const _Float16* rowbase, int c0) {
  const v8h lo = *(const v8h*)(rowbase + c0);
  const v8h hi = *(const v8h*)(rowbase + c0 + 8);
  v16h a;
  #pragma unroll
  for (int e = 0; e < 8; ++e) { a[e] = lo[e]; a[8 + e] = hi[e]; }
  return a;
}

__global__ __launch_bounds__(256)
void flash_attn_wmma_kernel(const float* __restrict__ Q,
                            const float* __restrict__ K,
                            const float* __restrict__ V,
                            float* __restrict__ O) {
  // double-buffered tiles (4 x 8KB = 32KB of the 320KB WGP LDS)
  __shared__ __align__(32) _Float16 ldsK[2][KTILE * Dn];   // [key][d] row-major f16
  __shared__ __align__(32) _Float16 ldsVT[2][Dn * KTILE];  // [d][key] (V transposed) f16

  const int tid  = threadIdx.x;
  const int wave = tid >> 5;
  const int lane = tid & 31;
  const int hl   = lane >> 4;   // lane half
  const int l16  = lane & 15;   // this lane's query column within the wave tile

  const int nqb   = Nn / QTILE;
  const int b     = blockIdx.x / nqb;
  const int qt    = blockIdx.x % nqb;
  const int qrow0 = qt * QTILE + wave * 16;

  const size_t bOff = (size_t)b * Nn * Dn;
  const float* Qb = Q + bOff;
  const float* Kb = K + bOff;
  const float* Vb = V + bOff;

  // ---- Q as B-layout fragments of Q^T (32x16): lane = query column, d contiguous ----
  v16h qb[2];
  {
    const float* qrow = Qb + (size_t)(qrow0 + l16) * Dn;
    #pragma unroll
    for (int f = 0; f < 2; ++f) {
      const int d0 = f * 32 + hl * 16;
      #pragma unroll
      for (int e = 0; e < 16; ++e) qb[f][e] = (_Float16)qrow[d0 + e];
    }
  }

  const v8f vzero = {};
  v8f o[4];                       // O^T col-tiles: vgpr v ~ d = t*16 + v + 8*hl, lane ~ query
  #pragma unroll
  for (int t = 0; t < 4; ++t) o[t] = vzero;
  float m = -3.0e38f, l = 0.0f;   // per-lane (per-query) softmax state

  // ---- cooperative staging helpers (register double buffer) ----
  f32x4 kreg[4], vreg[4];
  auto coop_load = [&](int kt) {
    #pragma unroll
    for (int i = 0; i < 4; ++i) {
      const int e4  = i * 256 + tid;       // 1024 float4 = 64x64 f32 tile
      const int row = e4 >> 4;
      const int c4  = (e4 & 15) << 2;
      const size_t g = (size_t)(kt + row) * Dn + c4;
      kreg[i] = *(const f32x4*)(Kb + g);
      vreg[i] = *(const f32x4*)(Vb + g);
    }
  };
  auto coop_store = [&](int buf) {
    #pragma unroll
    for (int i = 0; i < 4; ++i) {
      const int e4  = i * 256 + tid;
      const int row = e4 >> 4;
      const int c4  = (e4 & 15) << 2;
      v4h kh;
      #pragma unroll
      for (int j = 0; j < 4; ++j) kh[j] = (_Float16)kreg[i][j];
      *(v4h*)&ldsK[buf][row * Dn + c4] = kh;
      #pragma unroll
      for (int j = 0; j < 4; ++j)
        ldsVT[buf][(c4 + j) * KTILE + row] = (_Float16)vreg[i][j];
    }
  };

  coop_load(0);
  coop_store(0);
  __syncthreads();

  for (int it = 0; it < NTILES; ++it) {
    const int cur = it & 1;
    if (it + 1 < NTILES) coop_load((it + 1) * KTILE);  // loads fly under the WMMAs

    // ---- S^T tiles: st[j] = Ktile_rows(j) @ Q^T  (rows = keys, cols = queries) ----
    v8f s[4];
    #pragma unroll
    for (int j = 0; j < 4; ++j) {
      v8f acc = vzero;
      #pragma unroll
      for (int f = 0; f < 2; ++f) {
        const v16h ka = lds_afrag(&ldsK[cur][(j * 16 + l16) * Dn], f * 32 + hl * 16);
        acc = wmma_f16(ka, qb[f], acc);
      }
      s[j] = acc;
    }

    // ---- online softmax: per-lane query; only one xor-16 shuffle per reduction ----
    float tm = -3.0e38f;
    #pragma unroll
    for (int j = 0; j < 4; ++j)
      #pragma unroll
      for (int v = 0; v < 8; ++v) tm = fmaxf(tm, s[j][v]);
    tm = fmaxf(tm, __shfl_xor(tm, 16, 32));
    const float mnew = fmaxf(m, tm);
    const float sc   = __expf(m - mnew);
    m = mnew;
    float rs = 0.0f;
    #pragma unroll
    for (int j = 0; j < 4; ++j)
      #pragma unroll
      for (int v = 0; v < 8; ++v) {
        const float p = __expf(s[j][v] - mnew);
        s[j][v] = p;
        rs += p;
      }
    rs += __shfl_xor(rs, 16, 32);
    l = l * sc + rs;
    #pragma unroll
    for (int t = 0; t < 4; ++t) o[t] *= sc;

    // ---- P^T as B-layout fragments, built by register cross-half exchange ----
    // C layout of s[j]: vgpr v = key j*16+v (lanes 0-15) / j*16+v+8 (lanes 16-31), lane = query.
    // B frag f needs per lane: keys (2f+hl)*16 + 0..15 of its query column.
    v16h pb[2];
    #pragma unroll
    for (int f = 0; f < 2; ++f) {
      #pragma unroll
      for (int v = 0; v < 8; ++v) {
        const float mine = (hl == 0) ? s[2 * f][v]     : s[2 * f + 1][v];
        const float send = (hl == 0) ? s[2 * f + 1][v] : s[2 * f][v];
        const float oth  = __shfl_xor(send, 16, 32);
        pb[f][v]     = (_Float16)((hl == 0) ? mine : oth);  // k_rel = v
        pb[f][8 + v] = (_Float16)((hl == 0) ? oth  : mine); // k_rel = 8+v
      }
    }

    // ---- O^T tiles += V^T_rows(t) @ P^T ----
    #pragma unroll
    for (int t = 0; t < 4; ++t) {
      #pragma unroll
      for (int f = 0; f < 2; ++f) {
        const v16h va = lds_afrag(&ldsVT[cur][(t * 16 + l16) * KTILE], f * 32 + hl * 16);
        o[t] = wmma_f16(va, pb[f], o[t]);
      }
    }

    if (it + 1 < NTILES) coop_store(cur ^ 1);
    __syncthreads();
  }

  // ---- epilogue: per-lane normalize, contiguous 16B stores ----
  const float inv = 1.0f / l;
  float* orow = O + bOff + (size_t)(qrow0 + l16) * Dn + hl * 8;
  #pragma unroll
  for (int t = 0; t < 4; ++t) {
    f32x4 lo4, hi4;
    #pragma unroll
    for (int e = 0; e < 4; ++e) { lo4[e] = o[t][e] * inv; hi4[e] = o[t][4 + e] * inv; }
    *(f32x4*)(orow + t * 16)     = lo4;
    *(f32x4*)(orow + t * 16 + 4) = hi4;
  }
}

extern "C" void kernel_launch(void* const* d_in, const int* in_sizes, int n_in,
                              void* d_out, int out_size, void* d_ws, size_t ws_size,
                              hipStream_t stream) {
  const float* Q = (const float*)d_in[0];
  const float* K = (const float*)d_in[1];
  const float* V = (const float*)d_in[2];
  float* O = (float*)d_out;
  (void)in_sizes; (void)n_in; (void)out_size; (void)d_ws; (void)ws_size;
  dim3 grid(Bn * (Nn / QTILE));   // 8 batches * 32 query tiles = 256 blocks
  dim3 block(256);                // 8 wave32 waves
  flash_attn_wmma_kernel<<<grid, block, 0, stream>>>(Q, K, V, O);
}
